// GraphormerMultiheadAttention_58772332478794
// MI455X (gfx1250) — matrix operand order repaired and checked by reading, not compiled, not run on hardware
//
#include <hip/hip_runtime.h>
#include <hip/hip_bf16.h>

typedef __attribute__((ext_vector_type(16))) _Float16 v16h;
typedef __attribute__((ext_vector_type(8)))  _Float16 v8h;
typedef __attribute__((ext_vector_type(4)))  _Float16 v4h;
typedef __attribute__((ext_vector_type(8)))  float    v8f;

#define WMMA16(A, Bf, C) \
  __builtin_amdgcn_wmma_f32_16x16x32_f16(false, (A), false, (Bf), (short)0, (C), false, false)

static __device__ __forceinline__ v16h cat8(v8h lo, v8h hi) {
  return __builtin_shufflevector(lo, hi, 0,1,2,3,4,5,6,7,8,9,10,11,12,13,14,15);
}

// Problem constants (from the reference): B=2, T=S=2048, E=512, H=8, D=64
constexpr int kB = 2;
constexpr int kT = 2048;
constexpr int kS = 2048;
constexpr int kE = 512;
constexpr int kH = 8;
constexpr int kD = 64;

// ---------------------------------------------------------------------------
// QKV projection: Y = X @ W^T + b  (X fp32 [M=4096, 512], W fp32 [512,512])
// MODE 0: store f16 as [b*H+h][t][d]       (Q, K)
// MODE 1: store f16 transposed [b*H+h][d][s] (V, so PV B-fragments are contiguous)
// Block: 256 thr = 8 waves; tile 128(M) x 64(N); K staged 32 wide through LDS.
// ---------------------------------------------------------------------------
template<int MODE>
__global__ __launch_bounds__(256) void proj_qkv_kernel(
    const float* __restrict__ X, const float* __restrict__ W,
    const float* __restrict__ bias, _Float16* __restrict__ Y)
{
  __shared__ __align__(32) _Float16 Xs[128][48];  // 96B row stride (16B/32B aligned)
  __shared__ __align__(32) _Float16 Ws[64][48];

  const int tid  = threadIdx.x;
  const int wave = tid >> 5;
  const int lane = tid & 31;
  const int half = lane >> 4;
  const int l16  = lane & 15;
  const int m0   = blockIdx.x * 128;
  const int n0   = blockIdx.y * 64;

  const int lr = tid >> 3;        // 0..31 loader row
  const int lc = (tid & 7) * 4;   // 0..28 loader col (floats)

  v8f acc[4] = {};

  for (int k0 = 0; k0 < kE; k0 += 32) {
    #pragma unroll
    for (int rr = 0; rr < 128; rr += 32) {
      float4 f = *(const float4*)(X + (size_t)(m0 + rr + lr) * kE + k0 + lc);
      v4h h4 = { (_Float16)f.x, (_Float16)f.y, (_Float16)f.z, (_Float16)f.w };
      *(v4h*)&Xs[rr + lr][lc] = h4;
    }
    #pragma unroll
    for (int rr = 0; rr < 64; rr += 32) {
      float4 f = *(const float4*)(W + (size_t)(n0 + rr + lr) * kE + k0 + lc);
      v4h h4 = { (_Float16)f.x, (_Float16)f.y, (_Float16)f.z, (_Float16)f.w };
      *(v4h*)&Ws[rr + lr][lc] = h4;
    }
    __syncthreads();
    // A fragment: 16x32 f16, lane = row l16, chunks at k=8*half and 8*half+16
    v8h alo = *(const v8h*)&Xs[wave * 16 + l16][8 * half];
    v8h ahi = *(const v8h*)&Xs[wave * 16 + l16][8 * half + 16];
    v16h a = cat8(alo, ahi);
    #pragma unroll
    for (int j = 0; j < 4; ++j) {
      // B fragment: 32x16, lane = col l16, chunks at k=8*half and 8*half+16
      v16h bf = cat8(*(const v8h*)&Ws[j * 16 + l16][8 * half], *(const v8h*)&Ws[j * 16 + l16][8 * half + 16]);
      acc[j] = WMMA16(a, bf, acc[j]);
    }
    __syncthreads();
  }

  // Epilogue: C layout -> lane l holds col n=l16, rows m = mbase + v
  const int mbase = m0 + wave * 16 + 8 * half;
  #pragma unroll
  for (int j = 0; j < 4; ++j) {
    const int n  = n0 + j * 16 + l16;
    const float bv = bias[n];
    const int hh = n >> 6, dd = n & 63;
    if (MODE == 0) {
      #pragma unroll
      for (int v = 0; v < 8; ++v) {
        int m = mbase + v;
        int bi = m >> 11, t = m & 2047;
        Y[(((size_t)(bi * kH + hh)) * kT + t) * kD + dd] = (_Float16)(acc[j][v] + bv);
      }
    } else {
      v8h pk;
      #pragma unroll
      for (int v = 0; v < 8; ++v) pk[v] = (_Float16)(acc[j][v] + bv);
      int bi = mbase >> 11, s = mbase & 2047;   // 8 consecutive s -> one 16B store
      *(v8h*)&Y[(((size_t)(bi * kH + hh)) * kD + dd) * kS + s] = pk;
    }
  }
}

// ---------------------------------------------------------------------------
// Output projection: out = attn @ Wo^T + bo  (attn f16 [4096,512] -> out fp32)
// ---------------------------------------------------------------------------
__global__ __launch_bounds__(256) void proj_out_kernel(
    const _Float16* __restrict__ X, const float* __restrict__ W,
    const float* __restrict__ bias, float* __restrict__ Y)
{
  __shared__ __align__(32) _Float16 Xs[128][48];
  __shared__ __align__(32) _Float16 Ws[64][48];

  const int tid  = threadIdx.x;
  const int wave = tid >> 5;
  const int lane = tid & 31;
  const int half = lane >> 4;
  const int l16  = lane & 15;
  const int m0   = blockIdx.x * 128;
  const int n0   = blockIdx.y * 64;
  const int lr   = tid >> 3;
  const int lc   = (tid & 7) * 4;

  v8f acc[4] = {};

  for (int k0 = 0; k0 < kE; k0 += 32) {
    #pragma unroll
    for (int rr = 0; rr < 128; rr += 32) {
      v4h h4 = *(const v4h*)(X + (size_t)(m0 + rr + lr) * kE + k0 + lc);
      *(v4h*)&Xs[rr + lr][lc] = h4;
    }
    #pragma unroll
    for (int rr = 0; rr < 64; rr += 32) {
      float4 f = *(const float4*)(W + (size_t)(n0 + rr + lr) * kE + k0 + lc);
      v4h h4 = { (_Float16)f.x, (_Float16)f.y, (_Float16)f.z, (_Float16)f.w };
      *(v4h*)&Ws[rr + lr][lc] = h4;
    }
    __syncthreads();
    v8h alo = *(const v8h*)&Xs[wave * 16 + l16][8 * half];
    v8h ahi = *(const v8h*)&Xs[wave * 16 + l16][8 * half + 16];
    v16h a = cat8(alo, ahi);
    #pragma unroll
    for (int j = 0; j < 4; ++j) {
      v16h bf = cat8(*(const v8h*)&Ws[j * 16 + l16][8 * half], *(const v8h*)&Ws[j * 16 + l16][8 * half + 16]);
      acc[j] = WMMA16(a, bf, acc[j]);
    }
    __syncthreads();
  }

  const int mbase = m0 + wave * 16 + 8 * half;
  #pragma unroll
  for (int j = 0; j < 4; ++j) {
    const int n = n0 + j * 16 + l16;
    const float bv = bias[n];
    #pragma unroll
    for (int v = 0; v < 8; ++v) {
      Y[(size_t)(mbase + v) * kE + n] = acc[j][v] + bv;
    }
  }
}

// ---------------------------------------------------------------------------
// Fused flash-style attention per (b,h): each wave owns 16 query rows,
// streams S in chunks of 32 with online softmax. Biases/masks fused in f32.
// grid = (T/128, B*H), block = 256 (8 waves).
// ---------------------------------------------------------------------------
__global__ __launch_bounds__(256) void attn_kernel(
    const _Float16* __restrict__ q, const _Float16* __restrict__ k,
    const _Float16* __restrict__ vT,
    const float* __restrict__ amask, const float* __restrict__ sbias,
    const float* __restrict__ dbias, const unsigned char* __restrict__ kpm,
    _Float16* __restrict__ attn_out)
{
  __shared__ __align__(32) _Float16 Plds[8][16][40];  // per-wave 16x32 P tile, 80B rows

  const int tid  = threadIdx.x;
  const int wave = tid >> 5;
  const int lane = tid & 31;
  const int half = lane >> 4;
  const int l16  = lane & 15;
  const int bh   = blockIdx.y;
  const int b    = bh >> 3;
  const int h    = bh & 7;
  const int t0   = blockIdx.x * 128 + wave * 16;

  const _Float16* qh = q  + (size_t)bh * kT * kD;
  const _Float16* kh = k  + (size_t)bh * kS * kD;
  const _Float16* vh = vT + (size_t)bh * kD * kS;
  const float* sb = sbias + (size_t)b * kT * kS;
  const float* db = dbias + (size_t)b * kT * kS;
  const unsigned char* kp = kpm + (size_t)b * kS;

  // Q A-fragments for this wave's 16 rows (d 0..31 and 32..63), kept resident
  const int tr = t0 + l16;
  v16h aq0 = cat8(*(const v8h*)(qh + (size_t)tr * kD + 8 * half),
                  *(const v8h*)(qh + (size_t)tr * kD + 8 * half + 16));
  v16h aq1 = cat8(*(const v8h*)(qh + (size_t)tr * kD + 32 + 8 * half),
                  *(const v8h*)(qh + (size_t)tr * kD + 32 + 8 * half + 16));

  v8f o0 = {}, o1 = {}, o2 = {}, o3 = {};
  float mrow[8], lrow[8];
  #pragma unroll
  for (int v = 0; v < 8; ++v) { mrow[v] = -3.0e38f; lrow[v] = 0.f; }

  const float SCALE = 0.125f;            // D^-0.5, D=64
  const float L2E   = 1.44269504088896f;

  for (int s0 = 0; s0 < kS; s0 += 32) {
    if (s0 + 32 < kS) __builtin_prefetch(kh + (size_t)(s0 + 32) * kD, 0, 0);

    // ---- scores: two 16x16 C tiles covering columns [s0, s0+32)
    v8f sc0 = {}, sc1 = {};
    {
      int s = s0 + l16;
      v16h b0 = cat8(*(const v8h*)(kh + (size_t)s * kD + 8 * half), *(const v8h*)(kh + (size_t)s * kD + 8 * half + 16));        // d 0..31
      v16h b1 = cat8(*(const v8h*)(kh + (size_t)s * kD + 32 + 8 * half), *(const v8h*)(kh + (size_t)s * kD + 32 + 8 * half + 16));   // d 32..63
      sc0 = WMMA16(aq0, b0, sc0);
      sc0 = WMMA16(aq1, b1, sc0);
    }
    {
      int s = s0 + 16 + l16;
      v16h b0 = cat8(*(const v8h*)(kh + (size_t)s * kD + 8 * half), *(const v8h*)(kh + (size_t)s * kD + 8 * half + 16));
      v16h b1 = cat8(*(const v8h*)(kh + (size_t)s * kD + 32 + 8 * half), *(const v8h*)(kh + (size_t)s * kD + 32 + 8 * half + 16));
      sc1 = WMMA16(aq0, b0, sc1);
      sc1 = WMMA16(aq1, b1, sc1);
    }

    // ---- fuse scale + attn_mask + spatial + directional + key padding (f32)
    float x0[8], x1[8];
    {
      int s = s0 + l16;
      float pad = kp[s] ? -3.0e38f : 0.f;
      #pragma unroll
      for (int v = 0; v < 8; ++v) {
        int t = t0 + 8 * half + v;
        size_t off = (size_t)t * kS + s;
        x0[v] = sc0[v] * SCALE + amask[off] + sb[off] + db[off] + pad;
      }
    }
    {
      int s = s0 + 16 + l16;
      float pad = kp[s] ? -3.0e38f : 0.f;
      #pragma unroll
      for (int v = 0; v < 8; ++v) {
        int t = t0 + 8 * half + v;
        size_t off = (size_t)t * kS + s;
        x1[v] = sc1[v] * SCALE + amask[off] + sb[off] + db[off] + pad;
      }
    }

    // ---- online softmax (row stats live in 16-lane halves of C layout)
    #pragma unroll
    for (int v = 0; v < 8; ++v) {
      float cm = fmaxf(x0[v], x1[v]);
      cm = fmaxf(cm, __shfl_xor(cm, 1));
      cm = fmaxf(cm, __shfl_xor(cm, 2));
      cm = fmaxf(cm, __shfl_xor(cm, 4));
      cm = fmaxf(cm, __shfl_xor(cm, 8));
      float mn = fmaxf(mrow[v], cm);
      float fs = exp2f((mrow[v] - mn) * L2E);
      float p0 = exp2f((x0[v] - mn) * L2E);
      float p1 = exp2f((x1[v] - mn) * L2E);
      float ps = p0 + p1;
      ps += __shfl_xor(ps, 1);
      ps += __shfl_xor(ps, 2);
      ps += __shfl_xor(ps, 4);
      ps += __shfl_xor(ps, 8);
      lrow[v] = lrow[v] * fs + ps;
      mrow[v] = mn;
      o0[v] *= fs; o1[v] *= fs; o2[v] *= fs; o3[v] *= fs;
      // P tile (f16) into per-wave LDS: [row t_local][col s_local]
      Plds[wave][8 * half + v][l16]      = (_Float16)p0;
      Plds[wave][8 * half + v][16 + l16] = (_Float16)p1;
    }

    // per-wave C->A layout bounce through LDS; cross-lane, so wait all DS done
    asm volatile("s_wait_dscnt 0x0" ::: "memory");

    v16h ap = cat8(*(const v8h*)&Plds[wave][l16][16 * half],
                   *(const v8h*)&Plds[wave][l16][16 * half + 8]);

    // ---- O += P @ V  (V^T layout -> contiguous 32B B-fragments)
    {
      int d0 = l16;
      o0 = WMMA16(ap, *(const v16h*)(vh + (size_t)(d0)      * kS + s0 + 16 * half), o0);
      o1 = WMMA16(ap, *(const v16h*)(vh + (size_t)(16 + d0) * kS + s0 + 16 * half), o1);
      o2 = WMMA16(ap, *(const v16h*)(vh + (size_t)(32 + d0) * kS + s0 + 16 * half), o2);
      o3 = WMMA16(ap, *(const v16h*)(vh + (size_t)(48 + d0) * kS + s0 + 16 * half), o3);
    }
  }

  // ---- normalize and store per-head output, heads concatenated: [b][t][h*64+d]
  #pragma unroll
  for (int v = 0; v < 8; ++v) {
    float inv = 1.0f / lrow[v];
    int t = t0 + 8 * half + v;
    size_t rowoff = ((size_t)b * kT + t) * kE + h * kD;
    attn_out[rowoff + l16]      = (_Float16)(o0[v] * inv);
    attn_out[rowoff + 16 + l16] = (_Float16)(o1[v] * inv);
    attn_out[rowoff + 32 + l16] = (_Float16)(o2[v] * inv);
    attn_out[rowoff + 48 + l16] = (_Float16)(o3[v] * inv);
  }
}

// ---------------------------------------------------------------------------
extern "C" void kernel_launch(void* const* d_in, const int* in_sizes, int n_in,
                              void* d_out, int out_size, void* d_ws, size_t ws_size,
                              hipStream_t stream) {
  (void)in_sizes; (void)n_in; (void)out_size; (void)ws_size;
  const float* query = (const float*)d_in[0];
  const float* key   = (const float*)d_in[1];
  const float* value = (const float*)d_in[2];
  const float* sbias = (const float*)d_in[3];
  const float* dbias = (const float*)d_in[4];
  const unsigned char* kpm = (const unsigned char*)d_in[5];
  const float* amask = (const float*)d_in[6];
  const float* Wq = (const float*)d_in[7];
  const float* bq = (const float*)d_in[8];
  const float* Wk = (const float*)d_in[9];
  const float* bk = (const float*)d_in[10];
  const float* Wv = (const float*)d_in[11];
  const float* bv = (const float*)d_in[12];
  const float* Wo = (const float*)d_in[13];
  const float* bo = (const float*)d_in[14];
  float* out = (float*)d_out;

  // Workspace: q/k (4MB each, [bh][t|s][d] f16), vT (4MB, [bh][d][s] f16),
  // per-head attention output (4MB, [b][t][E] f16). Total 16MB.
  char* ws = (char*)d_ws;
  _Float16* qbuf = (_Float16*)(ws);
  _Float16* kbuf = (_Float16*)(ws + (size_t)4 * 1024 * 1024);
  _Float16* vbuf = (_Float16*)(ws + (size_t)8 * 1024 * 1024);
  _Float16* abuf = (_Float16*)(ws + (size_t)12 * 1024 * 1024);

  dim3 blk(256);
  dim3 gProj(32, 8);   // M=4096/128, N=512/64
  proj_qkv_kernel<0><<<gProj, blk, 0, stream>>>(query, Wq, bq, qbuf);
  proj_qkv_kernel<0><<<gProj, blk, 0, stream>>>(key,   Wk, bk, kbuf);
  proj_qkv_kernel<1><<<gProj, blk, 0, stream>>>(value, Wv, bv, vbuf);

  dim3 gAttn(16, 16);  // T/128, B*H
  attn_kernel<<<gAttn, blk, 0, stream>>>(qbuf, kbuf, vbuf, amask, sbias, dbias, kpm, abuf);

  proj_out_kernel<<<gProj, blk, 0, stream>>>(abuf, Wo, bo, out);
}
